// Upsample_62775241998778
// MI455X (gfx1250) — matrix operand order repaired and checked by reading, not compiled, not run on hardware
//
#include <hip/hip_runtime.h>

typedef __attribute__((ext_vector_type(16))) _Float16 v16h;
typedef __attribute__((ext_vector_type(8)))  _Float16 v8h;
typedef __attribute__((ext_vector_type(8)))  float    v8f;
typedef __attribute__((ext_vector_type(4)))  float    v4f;

#define C_CH     128
#define KSTEN    27
#define KK       (KSTEN * C_CH)       // 3456 reduction length
#define TCH      (KK / 32)            // 108 k-chunks of 32
#define M_TILE   32
#define ROWH     (KK + 8)             // padded LDS row in halves (3464)
#define NTHREADS 256
#define WP_HALVES (TCH * 8 * 32 * 16) // 442368 halves = 884736 B (B fragments)
// f16 copy of x lives in d_ws right after the B fragments (offset in halves):
#define XH_OFFSET WP_HALVES

// ---------------------------------------------------------------------------
// Prep 1: fp32 w[k][c][d] -> f16 B fragments for V_WMMA_F32_16X16X32_F16
// B operand, in this file's contiguous K order (matches A). Fragment (t, j): lane L holds column n = 16*j + (L&15),
// elements e=0..15 hold K = (L>>4)*16 + e of k-chunk t.
// ---------------------------------------------------------------------------
__global__ __launch_bounds__(NTHREADS) void prep_w_kernel(
    const float* __restrict__ w, _Float16* __restrict__ wp) {
  int idx = blockIdx.x * NTHREADS + threadIdx.x;
  if (idx >= WP_HALVES) return;
  int e    = idx & 15;
  int lane = (idx >> 4) & 31;
  int j    = (idx >> 9) & 7;
  int t    = idx >> 12;
  int h    = lane >> 4;
  int n    = j * 16 + (lane & 15);
  int kloc = h * 16 + e;            // K within the 32-chunk
  int koff = t >> 2;                // stencil offset 0..26
  int c    = (t & 3) * 32 + kloc;   // channel 0..127
  wp[idx] = (_Float16)w[(koff * C_CH + c) * C_CH + n];
}

// ---------------------------------------------------------------------------
// Prep 2: fp32 x[parent][c] -> f16 xh[parent][c] (one-time convert; all later
// gathers move f16 and need no VALU conversion).
// ---------------------------------------------------------------------------
__global__ __launch_bounds__(NTHREADS) void prep_x_kernel(
    const float* __restrict__ x, _Float16* __restrict__ xh, int ngroups) {
  int idx = blockIdx.x * NTHREADS + threadIdx.x;
  if (idx >= ngroups) return;                 // groups of 8 elements
  v4f f0 = ((const v4f*)x)[idx * 2];
  v4f f1 = ((const v4f*)x)[idx * 2 + 1];
  v8h o;
#pragma unroll
  for (int q = 0; q < 4; ++q) { o[q] = (_Float16)f0[q]; o[q + 4] = (_Float16)f1[q]; }
  ((v8h*)xh)[idx] = o;
}

// ---------------------------------------------------------------------------
// Main gather-GEMM: each block computes 32 children x 128 outputs.
// ---------------------------------------------------------------------------
__global__ __launch_bounds__(NTHREADS) void octconv_wmma_kernel(
    const _Float16* __restrict__ xh, const float* __restrict__ bias,
    const int* __restrict__ neigh, const _Float16* __restrict__ wp,
    float* __restrict__ out) {
  extern __shared__ _Float16 Al[];       // [M_TILE][ROWH] halves, 221696 B

  const int tid  = threadIdx.x;
  const int base = blockIdx.x * M_TILE;  // first child row of this block

  // ---- Stage A-tile via async copy: memory -> LDS, no VGPR round-trip. ----
  // Task i = (ko, m, cg), all power-of-two fields (no div/mod):
  //   cg = i & 15  (16B unit within row), m = (i>>4) & 31, ko = i >> 9.
  // A 16-thread group covers one contiguous 256B f16 neighbor row
  // (coalesced). Missing neighbor rows are zero-filled with ds_store_b128.
  for (int i = tid; i < KSTEN * M_TILE * 16; i += NTHREADS) {
    int cg = i & 15;
    int m  = (i >> 4) & (M_TILE - 1);
    int ko = i >> 9;                    // 0..26
    int ng = neigh[(base + m) * KSTEN + ko];
    unsigned dsth = (unsigned)(m * ROWH + ko * C_CH + cg * 8);   // half index
    if (ng >= 0) {
      unsigned ldsaddr = dsth * 2u;                              // byte offset
      unsigned long long gaddr =
          (unsigned long long)(xh + ((ng >> 3) * C_CH + cg * 8));
      asm volatile("global_load_async_to_lds_b128 %0, %1, off"
                   :: "v"(ldsaddr), "v"(gaddr) : "memory");
    } else {
      v8h z = {};
      *(v8h*)(Al + dsth) = z;
    }
  }
  asm volatile("s_wait_asynccnt 0x0" ::: "memory");
  __syncthreads();

  // ---- Compute: wave j owns N columns [16j,16j+16), both 16-row M subtiles.
  const int lane = tid & 31;
  const int j    = tid >> 5;      // 0..7
  const int h    = lane >> 4;     // half-wave
  const int col  = lane & 15;     // A row (lanes 0-15 / 16-31 both map M=0..15)

  v8f acc0 = {};
  v8f acc1 = {};

  union AFrag { v16h v; v8h half[2]; };

#pragma unroll 4
  for (int t = 0; t < TCH; ++t) {
    // B fragment: 32B contiguous per lane -> 2x global_load_b128 (L2-hot)
    v16h bf = *(const v16h*)(wp + t * 4096 + j * 512 + lane * 16);

    // A fragments from LDS, contiguous K order matching B fragments:
    // elements 0..15 <- K = t*32 + h*16 + e
    int abase = col * ROWH + t * 32 + h * 16;
    AFrag a0, a1;
    a0.half[0] = *(const v8h*)(Al + abase);
    a0.half[1] = *(const v8h*)(Al + abase + 8);
    a1.half[0] = *(const v8h*)(Al + abase + 16 * ROWH);
    a1.half[1] = *(const v8h*)(Al + abase + 16 * ROWH + 8);

    acc0 = __builtin_amdgcn_wmma_f32_16x16x32_f16(
        false, a0.v, false, bf, (short)0, acc0, false, false);
    acc1 = __builtin_amdgcn_wmma_f32_16x16x32_f16(
        false, a1.v, false, bf, (short)0, acc1, false, false);
  }

  // ---- Store with bias. C layout: VGPR r -> M = r + 8*h, N = lane&15. ----
  const int ncol = j * 16 + col;
  const float bv = bias[ncol];
#pragma unroll
  for (int r = 0; r < 8; ++r) {
    int m = r + 8 * h;
    out[(base + m) * C_CH + ncol]      = acc0[r] + bv;
    out[(base + 16 + m) * C_CH + ncol] = acc1[r] + bv;
  }
}

// ---------------------------------------------------------------------------
extern "C" void kernel_launch(void* const* d_in, const int* in_sizes, int n_in,
                              void* d_out, int out_size, void* d_ws, size_t ws_size,
                              hipStream_t stream) {
  const float* x     = (const float*)d_in[0];   // [N_PARENT, 128] f32
  const float* w     = (const float*)d_in[1];   // [27, 128, 128] f32
  const float* b     = (const float*)d_in[2];   // [128] f32
  const int*   neigh = (const int*)d_in[3];     // [N_CHILD, 27] i32
  float*       out   = (float*)d_out;           // [N_CHILD, 128] f32

  _Float16* wp = (_Float16*)d_ws;               // B fragments: 884736 B
  _Float16* xh = (_Float16*)d_ws + XH_OFFSET;   // f16 x: in_sizes[0]*2 B

  const int n_child  = in_sizes[3] / KSTEN;     // 131072
  const int xgroups  = in_sizes[0] / 8;         // 8-element convert groups

  prep_w_kernel<<<(WP_HALVES + NTHREADS - 1) / NTHREADS, NTHREADS, 0, stream>>>(w, wp);
  prep_x_kernel<<<(xgroups + NTHREADS - 1) / NTHREADS, NTHREADS, 0, stream>>>(x, xh, xgroups);

  const size_t shmem = (size_t)M_TILE * ROWH * sizeof(_Float16);  // 221696 B
  octconv_wmma_kernel<<<n_child / M_TILE, NTHREADS, shmem, stream>>>(
      xh, b, neigh, wp, out);
}
